// PatchCoherentLoss_66941360275612
// MI455X (gfx1250) — matrix operand order repaired and checked by reading, not compiled, not run on hardware
//
#include <hip/hip_runtime.h>

typedef __attribute__((ext_vector_type(16))) _Float16 v16h;
typedef __attribute__((ext_vector_type(8)))  _Float16 v8h;
typedef __attribute__((ext_vector_type(8)))  float    v8f;

constexpr int NP    = 8100;   // 90*90 valid patches per image
constexpr int NPAD  = 8192;   // padded rows: multiple of 128 (block i-tile)
constexpr int KP    = 160;    // K padded: 147 -> 5*32
constexpr int DREAL = 147;
constexpr int BATCH = 2;
constexpr int LDSW  = 168;    // LDS row stride (halves), padded vs 160

// ---- order-preserving float <-> uint key (for atomicMin on floats) ----
__device__ __forceinline__ unsigned fkey(float f) {
  unsigned u = __float_as_uint(f);
  return (u & 0x80000000u) ? ~u : (u | 0x80000000u);
}
__device__ __forceinline__ float funkey(unsigned k) {
  return __uint_as_float((k & 0x80000000u) ? (k ^ 0x80000000u) : ~k);
}

__global__ void init_keys(unsigned* keys, int n) {
  int i = blockIdx.x * blockDim.x + threadIdx.x;
  if (i < n) keys[i] = 0xFFFFFFFFu;
}

// patch extraction + f32->f16 cast, zero-padded to (NPAD x KP)
__global__ void prep_patches(const float* __restrict__ src, _Float16* __restrict__ dst) {
  int idx = blockIdx.x * blockDim.x + threadIdx.x;
  const int total = BATCH * NPAD * KP;
  if (idx >= total) return;
  int k   = idx % KP;
  int row = idx / KP;
  int p   = row % NPAD;
  int b   = row / NPAD;
  _Float16 v = (_Float16)0.0f;
  if (p < NP && k < DREAL) {
    int r = p / 90, c = p % 90;
    int ch = k / 49, rem = k % 49;
    int dr = rem / 7, dc = rem % 7;
    v = (_Float16)src[(((b * 3 + ch) * 96) + (r + dr)) * 96 + (c + dc)];
  }
  dst[idx] = v;
}

// per-row squared norms (one wave per row, f32 accumulate)
__global__ void row_norms(const _Float16* __restrict__ patches, float* __restrict__ norms) {
  int row  = blockIdx.x * 8 + (threadIdx.x >> 5);
  int lane = threadIdx.x & 31;
  const _Float16* pr = patches + (size_t)row * KP;
  float s = 0.f;
  #pragma unroll
  for (int j = 0; j < 5; ++j) {
    float v = (float)pr[lane * 5 + j];
    s += v * v;
  }
  #pragma unroll
  for (int m = 1; m < 32; m <<= 1) s += __shfl_xor(s, m, 32);
  if (lane == 0) norms[row] = s;
}

__device__ __forceinline__ v16h frag16(const v8h& lo, const v8h& hi) {
  v16h f;
  #pragma unroll
  for (int j = 0; j < 8; ++j) { f[j] = lo[j]; f[j + 8] = hi[j]; }
  return f;
}

// Block tile: 64 t-rows x 128 i-cols; 8 waves in 2x4; each wave a 32x32 tile
// with 2x2 WMMA accumulators (each A/B fragment feeds two WMMAs).
// PASS 1: m_t = min_i dist[t,i]            (atomicMin into mkeys)
// PASS 2: v_i = min_t dist[t,i]/(m_t + a)  (atomicMin into vkeys)
template<int PASS>
__global__ __launch_bounds__(256)
void coherence_gemm(const _Float16* __restrict__ Xp, const _Float16* __restrict__ Yp,
                    const float* __restrict__ x2, const float* __restrict__ y2,
                    unsigned* __restrict__ mkeys, unsigned* __restrict__ vkeys) {
  __shared__ __align__(16) _Float16 Ys[64 * LDSW];
  __shared__ __align__(16) _Float16 Xs[128 * LDSW];

  const int b  = blockIdx.z;
  const int I0 = blockIdx.x * 128;  // input-patch (column) tile base
  const int T0 = blockIdx.y * 64;   // target-patch (row) tile base

  const _Float16* Xg = Xp + (size_t)b * NPAD * KP;
  const _Float16* Yg = Yp + (size_t)b * NPAD * KP;
  const float* x2g = x2 + b * NPAD;
  const float* y2g = y2 + b * NPAD;
  unsigned* mk = mkeys + b * NPAD;
  unsigned* vk = vkeys + b * NPAD;

  const int tid = threadIdx.x;

  // stage Y(64x160) and X(128x160) into LDS in 16B chunks (20 chunks/row)
  for (int idx = tid; idx < 64 * 20; idx += 256) {
    int row = idx / 20, seg = idx % 20;
    *(v8h*)&Ys[row * LDSW + seg * 8] = *(const v8h*)(Yg + (size_t)(T0 + row) * KP + seg * 8);
  }
  for (int idx = tid; idx < 128 * 20; idx += 256) {
    int row = idx / 20, seg = idx % 20;
    *(v8h*)&Xs[row * LDSW + seg * 8] = *(const v8h*)(Xg + (size_t)(I0 + row) * KP + seg * 8);
  }
  __syncthreads();

  const int wave = tid >> 5;
  const int lane = tid & 31;
  const int wrow = wave >> 2;   // 0..1 : which 32-row slab
  const int wcol = wave & 3;    // 0..3 : which 32-col slab
  const int hsel = lane >> 4;   // half-wave select
  const int lrow = lane & 15;

  const _Float16* ya0 = &Ys[(32 * wrow + lrow) * LDSW];        // t rows [0,16)
  const _Float16* ya1 = ya0 + 16 * LDSW;                       // t rows [16,32)
  const _Float16* xb0 = &Xs[(32 * wcol + lrow) * LDSW];        // i cols [0,16)
  const _Float16* xb1 = xb0 + 16 * LDSW;                       // i cols [16,32)

  v8f acc00 = {}, acc01 = {}, acc10 = {}, acc11 = {};
  #pragma unroll
  for (int kk = 0; kk < 5; ++kk) {
    const int kb = kk * 32;
    // A fragments (16x32 f16): lane = row m; halves [8*hsel..+7] and [+16..+7]
    v16h a0 = frag16(*(const v8h*)(ya0 + kb + 8 * hsel),
                     *(const v8h*)(ya0 + kb + 8 * hsel + 16));
    v16h a1 = frag16(*(const v8h*)(ya1 + kb + 8 * hsel),
                     *(const v8h*)(ya1 + kb + 8 * hsel + 16));
    // B fragments (32x16 f16): lane = column n; halves [8*hsel..+7] and [+16..+7]
    v16h b0 = frag16(*(const v8h*)(xb0 + kb + 8 * hsel),
                     *(const v8h*)(xb0 + kb + 8 * hsel + 16));
    v16h b1 = frag16(*(const v8h*)(xb1 + kb + 8 * hsel),
                     *(const v8h*)(xb1 + kb + 8 * hsel + 16));
    acc00 = __builtin_amdgcn_wmma_f32_16x16x32_f16(false, a0, false, b0, (short)0, acc00, false, false);
    acc01 = __builtin_amdgcn_wmma_f32_16x16x32_f16(false, a0, false, b1, (short)0, acc01, false, false);
    acc10 = __builtin_amdgcn_wmma_f32_16x16x32_f16(false, a1, false, b0, (short)0, acc10, false, false);
    acc11 = __builtin_amdgcn_wmma_f32_16x16x32_f16(false, a1, false, b1, (short)0, acc11, false, false);
  }

  // epilogue: C/D layout — VGPR r holds row (r + 8*hsel), lane lrow holds column
  const float INF = __builtin_inff();
  const int T0w = T0 + 32 * wrow;
  const int I0w = I0 + 32 * wcol;
  const int i0 = I0w + lrow;        // columns of acc*0
  const int i1 = I0w + 16 + lrow;   // columns of acc*1
  const bool i0ok = (i0 < NP), i1ok = (i1 < NP);
  const float x0 = i0ok ? x2g[i0] : 0.f;
  const float x1 = i1ok ? x2g[i1] : 0.f;

  if (PASS == 1) {
    // acc(t0) pairs {acc00,acc01} and {acc10,acc11} share t rows; min columns first
    #pragma unroll
    for (int tt = 0; tt < 2; ++tt) {
      const v8f& aL = tt ? acc10 : acc00;
      const v8f& aR = tt ? acc11 : acc01;
      #pragma unroll
      for (int r = 0; r < 8; ++r) {
        int t = T0w + 16 * tt + r + 8 * hsel;
        bool tok = (t < NP);
        float d0 = (i0ok && tok) ? (y2g[t] + x0 - 2.f * aL[r]) * (1.f / DREAL) : INF;
        float d1 = (i1ok && tok) ? (y2g[t] + x1 - 2.f * aR[r]) * (1.f / DREAL) : INF;
        float d = fminf(d0, d1);
        #pragma unroll
        for (int s = 1; s < 16; s <<= 1) d = fminf(d, __shfl_xor(d, s, 32));
        if (lrow == 0 && tok) atomicMin(&mk[t], fkey(d));
      }
    }
  } else {
    // acc(c0) pairs {acc00,acc10} and {acc01,acc11} share i columns
    #pragma unroll
    for (int cc = 0; cc < 2; ++cc) {
      const v8f& aT = cc ? acc01 : acc11;   // note: pick pair sharing column cc
      const v8f& aB = cc ? acc11 : acc01;   // placeholders fixed below
      (void)aT; (void)aB;
      const v8f& top = cc ? acc01 : acc00;  // rows [0,16) of wave tile
      const v8f& bot = cc ? acc11 : acc10;  // rows [16,32)
      const int i = cc ? i1 : i0;
      const bool iok = cc ? i1ok : i0ok;
      const float xin = cc ? x1 : x0;
      float vmin = INF;
      #pragma unroll
      for (int r = 0; r < 8; ++r) {
        int t0 = T0w + r + 8 * hsel;
        int t1 = T0w + 16 + r + 8 * hsel;
        if (iok && t0 < NP) {
          float d = (y2g[t0] + xin - 2.f * top[r]) * (1.f / DREAL);
          vmin = fminf(vmin, d / (funkey(mk[t0]) + 0.05f));
        }
        if (iok && t1 < NP) {
          float d = (y2g[t1] + xin - 2.f * bot[r]) * (1.f / DREAL);
          vmin = fminf(vmin, d / (funkey(mk[t1]) + 0.05f));
        }
      }
      vmin = fminf(vmin, __shfl_xor(vmin, 16, 32));  // combine rows 0-7 / 8-15
      if (hsel == 0 && iok) atomicMin(&vk[i], fkey(vmin));
    }
  }
}

// mean of per-image means == global mean (equal counts per image)
__global__ void finalize(const unsigned* __restrict__ vkeys, float* __restrict__ out) {
  __shared__ float red[256];
  float s = 0.f;
  for (int idx = threadIdx.x; idx < BATCH * NP; idx += 256) {
    int b = idx / NP, i = idx % NP;
    s += funkey(vkeys[b * NPAD + i]);
  }
  red[threadIdx.x] = s;
  __syncthreads();
  for (int w = 128; w > 0; w >>= 1) {
    if (threadIdx.x < w) red[threadIdx.x] += red[threadIdx.x + w];
    __syncthreads();
  }
  if (threadIdx.x == 0) out[0] = red[0] / (float)(BATCH * NP);
}

extern "C" void kernel_launch(void* const* d_in, const int* in_sizes, int n_in,
                              void* d_out, int out_size, void* d_ws, size_t ws_size,
                              hipStream_t stream) {
  (void)in_sizes; (void)n_in; (void)out_size; (void)ws_size;
  const float* x = (const float*)d_in[0];
  const float* y = (const float*)d_in[1];

  // workspace layout (all offsets 16B-aligned)
  const size_t patchHalves = (size_t)BATCH * NPAD * KP;
  _Float16* Xp = (_Float16*)d_ws;
  _Float16* Yp = Xp + patchHalves;
  float*    x2 = (float*)(Yp + patchHalves);
  float*    y2 = x2 + BATCH * NPAD;
  unsigned* mk = (unsigned*)(y2 + BATCH * NPAD);
  unsigned* vk = mk + BATCH * NPAD;   // contiguous after mk

  {
    int nkeys = 2 * BATCH * NPAD;     // mk + vk in one shot
    init_keys<<<(nkeys + 255) / 256, 256, 0, stream>>>(mk, nkeys);
  }
  {
    int total  = BATCH * NPAD * KP;
    int blocks = (total + 255) / 256;
    prep_patches<<<blocks, 256, 0, stream>>>(x, Xp);
    prep_patches<<<blocks, 256, 0, stream>>>(y, Yp);
  }
  {
    int rows = BATCH * NPAD;          // multiple of 8
    row_norms<<<rows / 8, 256, 0, stream>>>(Xp, x2);
    row_norms<<<rows / 8, 256, 0, stream>>>(Yp, y2);
  }
  dim3 grid(NPAD / 128, NPAD / 64, BATCH);
  coherence_gemm<1><<<grid, 256, 0, stream>>>(Xp, Yp, x2, y2, mk, vk);
  coherence_gemm<2><<<grid, 256, 0, stream>>>(Xp, Yp, x2, y2, mk, vk);
  finalize<<<1, 256, 0, stream>>>(vk, (float*)d_out);
}
